// SOGNN2_47313359733017
// MI455X (gfx1250) — matrix operand order repaired and checked by reading, not compiled, not run on hardware
//
#include <hip/hip_runtime.h>
#include <hip/hip_bf16.h>

typedef __attribute__((ext_vector_type(16))) _Float16 v16h;
typedef __attribute__((ext_vector_type(8)))  float    v8f;

// ---------------------------------------------------------------------------
// Max over groups of 4 lanes via DPP quad_perm (no LDS traffic, no dscnt).
// ---------------------------------------------------------------------------
static __device__ __forceinline__ float pool4_max(float v) {
  int x = __builtin_amdgcn_mov_dpp(__float_as_int(v), 0xB1, 0xF, 0xF, true); // xor1
  v = fmaxf(v, __int_as_float(x));
  x = __builtin_amdgcn_mov_dpp(__float_as_int(v), 0x4E, 0xF, 0xF, true);     // xor2
  return fmaxf(v, __int_as_float(x));
}

// ---------------------------------------------------------------------------
// Async global->LDS copy, 16B per lane (GLOBAL_LOAD_ASYNC_TO_LDS_B128,
// tracked by ASYNCcnt).  LDS operand is the byte address of the destination.
// ---------------------------------------------------------------------------
static __device__ __forceinline__ void async_copy16(const void* gsrc, void* ldst) {
  unsigned loff = (unsigned)(uintptr_t)ldst;
  unsigned long long ga = (unsigned long long)(uintptr_t)gsrc;
  asm volatile("global_load_async_to_lds_b128 %0, %1, off"
               :: "v"(loff), "v"(ga) : "memory");
}
static __device__ __forceinline__ void async_wait0() {
  asm volatile("s_wait_asynccnt 0x0" ::: "memory");
}

// ---------------------------------------------------------------------------
// A fragment: 16x32 f16, row-major source, contiguous k order: lane half h holds k = 16h + s.
// Lane layout makes each lane's 8 dwords two contiguous 16B runs -> 2x b128.
// ---------------------------------------------------------------------------
static __device__ __forceinline__ v16h load_a_frag16(const _Float16* __restrict__ M,
                                                     int ld, int m, int k0) {
  const int half = (threadIdx.x & 31) >> 4;
  const uint4* p = (const uint4*)(M + (size_t)m * ld + k0);
  union { v16h v; uint4 q[2]; } f;
  f.q[0] = p[2 * half];
  f.q[1] = p[2 * half + 1];
  return f.v;
}

// ---------------------------------------------------------------------------
// B fragment from pre-swizzled weights: fragment-contiguous, 2x b128 per lane.
// Swizzled layout (dwords): (((k0/32)*N + n)*2 + half)*8 + j
// holds f16 pair (k = k0 + half*16 + 2j, n), (k+1, n).
// ---------------------------------------------------------------------------
static __device__ __forceinline__ v16h load_b_swz(const unsigned* __restrict__ W,
                                                  int N, int k0, int n0) {
  const int lane = threadIdx.x & 31;
  const int half = lane >> 4;
  const int n = n0 + (lane & 15);
  const size_t idx = ((((size_t)(k0 >> 5) * N + n) * 2) + half) * 2;  // in uint4 units
  const uint4* p = (const uint4*)W + idx;
  union { v16h v; uint4 q[2]; } f;
  f.q[0] = p[0];
  f.q[1] = p[1];
  return f.v;
}

// ---------------------------------------------------------------------------
// Prep 1: conv filter matrices -> f16 row-major (K zero-padded for conv1)
// ---------------------------------------------------------------------------
__global__ void prep_conv(const float* __restrict__ w1, const float* __restrict__ w2,
                          const float* __restrict__ w3,
                          _Float16* __restrict__ A1, _Float16* __restrict__ A2,
                          _Float16* __restrict__ A3) {
  const int total = 1024 + 10240 + 40960;
  for (int i = blockIdx.x * blockDim.x + threadIdx.x; i < total; i += gridDim.x * blockDim.x) {
    int idx = i;
    if (idx < 1024) {
      int o = idx >> 5, k = idx & 31;
      A1[idx] = (_Float16)(k < 25 ? w1[o * 25 + k] : 0.0f);
      continue;
    }
    idx -= 1024;
    if (idx < 10240) { A2[idx] = (_Float16)w2[idx]; continue; }
    idx -= 10240;
    A3[idx] = (_Float16)w3[idx];
  }
}

// ---------------------------------------------------------------------------
// Prep 2: swizzle a KxN f32 matrix into fragment-contiguous f16 dwords
// ---------------------------------------------------------------------------
__global__ void prep_swz(const float* __restrict__ w, unsigned* __restrict__ out,
                         int K, int N) {
  const int total = (K * N) >> 1;
  for (int d = blockIdx.x * blockDim.x + threadIdx.x; d < total; d += gridDim.x * blockDim.x) {
    int j = d & 7;
    int h = (d >> 3) & 1;
    int n = (d >> 4) % N;
    int kb = d / (16 * N);
    int k = kb * 32 + h * 16 + 2 * j;
    union { unsigned u; _Float16 e[2]; } p;
    p.e[0] = (_Float16)w[(size_t)k * N + n];
    p.e[1] = (_Float16)w[(size_t)(k + 1) * N + n];
    out[d] = p.u;
  }
}

// ---------------------------------------------------------------------------
// conv1: (5x264 f32) -> 5x5 conv (32ch, N=260, K=25 pad 32) -> relu -> pool4
// Unconditional LDS gathers: pad rows (k=25..31) read zeroed LDS region;
// junk columns (n>=260) are finite and never stored (p<65 guard).
// ---------------------------------------------------------------------------
#define C1_LDS (7 * 264 + 280)
__global__ __launch_bounds__(256) void conv1_kernel(const float* __restrict__ x,
                                                    const _Float16* __restrict__ A1,
                                                    const float* __restrict__ bias,
                                                    _Float16* __restrict__ h1) {
  __shared__ _Float16 sIn[C1_LDS];
  const int row = blockIdx.x;
  const float* xr = x + (size_t)row * 1320;
  __builtin_prefetch(A1, 0, 3);
  for (int i = threadIdx.x; i < 1320; i += 256) sIn[i] = (_Float16)xr[i];
  for (int i = 1320 + threadIdx.x; i < C1_LDS; i += 256) sIn[i] = (_Float16)0.0f;
  __syncthreads();

  const int lane = threadIdx.x & 31, wave = threadIdx.x >> 5;
  const int half = lane >> 4, lm = lane & 15;
  for (int tile = wave; tile < 34; tile += 8) {      // 2 M-tiles x 17 N-tiles
    const int mt = tile / 17, nt = tile % 17;
    const int m0 = mt * 16, n0 = nt * 16;
    const int n = n0 + lm;
    v16h a = load_a_frag16(A1, 32, m0 + lm, 0);
    union { v16h v; _Float16 e[16]; } bf;
#pragma unroll
    for (int j = 0; j < 8; ++j) {
#pragma unroll
      for (int t = 0; t < 2; ++t) {
        const int kk = (half ? 16 : 0) + 2 * j + t;  // compile-time per branch
        const int kr = kk / 5, kc = kk % 5;
        bf.e[2 * j + t] = sIn[kr * 264 + n + kc];
      }
    }
    v8f acc = {};
    acc = __builtin_amdgcn_wmma_f32_16x16x32_f16(false, a, false, bf.v, (short)0, acc,
                                                 false, false);
#pragma unroll
    for (int r = 0; r < 8; ++r) {
      int M = m0 + r + (half ? 8 : 0);
      float v = acc[r] + bias[M];
      v = v > 0.f ? v : 0.f;
      v = pool4_max(v);
      if ((lane & 3) == 0) {
        int p = n >> 2;
        if (p < 65) h1[((size_t)row * 32 + M) * 65 + p] = (_Float16)v;
      }
    }
  }
}

// ---------------------------------------------------------------------------
// conv2: (32,65) f16 -> 1x5 conv (64ch, N=61, K=160) -> relu -> pool4 -> (64,15)
// LDS staged via GLOBAL_LOAD_ASYNC_TO_LDS_B128 (ASYNCcnt).
// ---------------------------------------------------------------------------
#define C2_LDS (32 * 65 + 16)
__global__ __launch_bounds__(256) void conv2_kernel(const _Float16* __restrict__ h1,
                                                    const _Float16* __restrict__ A2,
                                                    const float* __restrict__ bias,
                                                    _Float16* __restrict__ h2) {
  __shared__ _Float16 sIn[C2_LDS];
  const int row = blockIdx.x;
  const _Float16* hr = h1 + (size_t)row * 2080;
  __builtin_prefetch(A2, 0, 3);
  for (int i = threadIdx.x; i < 260; i += 256)       // 2080 f16 = 260 x 16B
    async_copy16(hr + i * 8, &sIn[i * 8]);
  for (int i = 2080 + threadIdx.x; i < C2_LDS; i += 256) sIn[i] = (_Float16)0.0f;
  async_wait0();
  __syncthreads();

  const int lane = threadIdx.x & 31, wave = threadIdx.x >> 5;
  const int half = lane >> 4, lm = lane & 15;
  for (int tile = wave; tile < 16; tile += 8) {      // 4 M-tiles x 4 N-tiles
    const int mt = tile >> 2, nt = tile & 3;
    const int m0 = mt * 16, n0 = nt * 16;
    const int n = n0 + lm;
    v8f acc = {};
#pragma unroll
    for (int k0 = 0; k0 < 160; k0 += 32) {
      v16h a = load_a_frag16(A2, 160, m0 + lm, k0);
      union { v16h v; _Float16 e[16]; } bf;
#pragma unroll
      for (int j = 0; j < 8; ++j) {
#pragma unroll
        for (int t = 0; t < 2; ++t) {
          const int kk = k0 + (half ? 16 : 0) + 2 * j + t;  // constant after unroll
          const int ic = kk / 5, kc = kk % 5;
          bf.e[2 * j + t] = sIn[ic * 65 + n + kc];
        }
      }
      acc = __builtin_amdgcn_wmma_f32_16x16x32_f16(false, a, false, bf.v, (short)0, acc,
                                                   false, false);
    }
#pragma unroll
    for (int r = 0; r < 8; ++r) {
      int M = m0 + r + (half ? 8 : 0);
      float v = acc[r] + bias[M];
      v = v > 0.f ? v : 0.f;
      v = pool4_max(v);
      if ((lane & 3) == 0) {
        int p = n >> 2;
        if (p < 15) h2[((size_t)row * 64 + M) * 15 + p] = (_Float16)v;
      }
    }
  }
}

// ---------------------------------------------------------------------------
// conv3: (64,15) f16 -> 1x5 conv (128ch, N=11, K=320) -> relu -> pool4 -> (128,2)
// ---------------------------------------------------------------------------
#define C3_LDS (64 * 15 + 24)
__global__ __launch_bounds__(256) void conv3_kernel(const _Float16* __restrict__ h2,
                                                    const _Float16* __restrict__ A3,
                                                    const float* __restrict__ bias,
                                                    _Float16* __restrict__ h3) {
  __shared__ _Float16 sIn[C3_LDS];
  const int row = blockIdx.x;
  const _Float16* hr = h2 + (size_t)row * 960;
  __builtin_prefetch(A3, 0, 3);
  if (threadIdx.x < 120)                             // 960 f16 = 120 x 16B
    async_copy16(hr + threadIdx.x * 8, &sIn[threadIdx.x * 8]);
  for (int i = 960 + threadIdx.x; i < C3_LDS; i += 256) sIn[i] = (_Float16)0.0f;
  async_wait0();
  __syncthreads();

  const int lane = threadIdx.x & 31, wave = threadIdx.x >> 5;
  const int half = lane >> 4, lm = lane & 15;
  {
    const int m0 = wave * 16;                        // 8 M-tiles x 1 N-tile
    const int n = lm;
    v8f acc = {};
#pragma unroll
    for (int k0 = 0; k0 < 320; k0 += 32) {
      v16h a = load_a_frag16(A3, 320, m0 + lm, k0);
      union { v16h v; _Float16 e[16]; } bf;
#pragma unroll
      for (int j = 0; j < 8; ++j) {
#pragma unroll
        for (int t = 0; t < 2; ++t) {
          const int kk = k0 + (half ? 16 : 0) + 2 * j + t;
          const int ic = kk / 5, kc = kk % 5;
          bf.e[2 * j + t] = sIn[ic * 15 + n + kc];
        }
      }
      acc = __builtin_amdgcn_wmma_f32_16x16x32_f16(false, a, false, bf.v, (short)0, acc,
                                                   false, false);
    }
#pragma unroll
    for (int r = 0; r < 8; ++r) {
      int M = m0 + r + (half ? 8 : 0);
      float v = acc[r] + bias[M];
      v = v > 0.f ? v : 0.f;
      v = pool4_max(v);
      if ((lane & 3) == 0) {
        int p = n >> 2;
        if (p < 2) h3[((size_t)row * 128 + M) * 2 + p] = (_Float16)v;
      }
    }
  }
}

// ---------------------------------------------------------------------------
// SOGC (templated on F): WMMA for feat@w_bn (62xFx64) and feat@w_g (62xFx32),
// f32 VALU for the tiny 62x62 adjacency pipeline.
// ---------------------------------------------------------------------------
template <int F>
__global__ __launch_bounds__(256) void sogc_kernel(const _Float16* __restrict__ feat,
                                                   const unsigned* __restrict__ wbn,
                                                   const float* __restrict__ bbn,
                                                   const unsigned* __restrict__ wg,
                                                   const float* __restrict__ bg,
                                                   float* __restrict__ feats_out, int stage) {
  __shared__ float sXa[62 * 64];
  __shared__ float sG[62 * 32];
  __shared__ float sAdj[62 * 62];
  __shared__ float sDeg[62];

  const int b = blockIdx.x;
  const _Float16* fb = feat + (size_t)b * 62 * F;
  const int lane = threadIdx.x & 31, wave = threadIdx.x >> 5;
  const int half = lane >> 4, lm = lane & 15;

  // 16 xa tiles (4x4) + 8 G tiles (4x2) = 24 tiles, 3 per wave
  for (int tile = wave; tile < 24; tile += 8) {
    const bool isXa = tile < 16;
    int mt, nt, N;
    const unsigned* W;
    if (isXa) { mt = tile >> 2; nt = tile & 3; W = wbn; N = 64; }
    else      { int t2 = tile - 16; mt = t2 >> 1; nt = t2 & 1; W = wg; N = 32; }
    const int m0 = mt * 16, n0 = nt * 16;
    const int mA = m0 + lm > 61 ? 61 : m0 + lm;      // clamp; rows 62/63 discarded
    v8f acc = {};
    for (int k0 = 0; k0 < F; k0 += 32) {
      v16h a  = load_a_frag16(fb, F, mA, k0);
      v16h bv = load_b_swz(W, N, k0, n0);
      acc = __builtin_amdgcn_wmma_f32_16x16x32_f16(false, a, false, bv, (short)0, acc,
                                                   false, false);
    }
    const int n = n0 + lm;
    if (isXa) {
      float bias = bbn[n];
#pragma unroll
      for (int r = 0; r < 8; ++r) {
        int M = m0 + r + (half ? 8 : 0);
        if (M < 62) sXa[M * 64 + n] = tanhf(acc[r] + bias);
      }
    } else {
#pragma unroll
      for (int r = 0; r < 8; ++r) {
        int M = m0 + r + (half ? 8 : 0);
        if (M < 62) sG[M * 32 + n] = acc[r];
      }
    }
  }
  __syncthreads();

  // Per-row adjacency: S = xa@xa^T, softmax, top-10 mask, diag, degree
  if (threadIdx.x < 62) {
    const int r = threadIdx.x;
    float mx = -1e30f;
    for (int j = 0; j < 62; ++j) {
      float s = 0.f;
      for (int kk = 0; kk < 64; ++kk) s += sXa[r * 64 + kk] * sXa[j * 64 + kk];
      sAdj[r * 62 + j] = s;
      mx = fmaxf(mx, s);
    }
    float sum = 0.f;
    for (int j = 0; j < 62; ++j) {
      float e = __expf(sAdj[r * 62 + j] - mx);
      sAdj[r * 62 + j] = e;
      sum += e;
    }
    float inv = 1.f / sum;
    for (int j = 0; j < 62; ++j) sAdj[r * 62 + j] *= inv;
    unsigned long long msk = 0ull;
    for (int k = 0; k < 10; ++k) {
      float best = -1.f; int bj = 0;
      for (int j = 0; j < 62; ++j) {
        if (!((msk >> j) & 1ull)) {
          float v = sAdj[r * 62 + j];
          if (v > best) { best = v; bj = j; }
        }
      }
      msk |= 1ull << bj;
    }
    for (int j = 0; j < 62; ++j)
      if (!((msk >> j) & 1ull)) sAdj[r * 62 + j] = 0.f;
    sAdj[r * 62 + r] = 1.f;
    float rs = 0.f;
    for (int j = 0; j < 62; ++j) rs += sAdj[r * 62 + j];
    rs = rs < 1.f ? 1.f : rs;
    sDeg[r] = rsqrtf(rs);
  }
  __syncthreads();
  if (threadIdx.x < 62) {
    const int r = threadIdx.x;
    const float dr = sDeg[r];
    for (int j = 0; j < 62; ++j) sAdj[r * 62 + j] *= dr * sDeg[j];
  }
  __syncthreads();

  // out = relu(adjn @ G + b_g)
  for (int idx = threadIdx.x; idx < 62 * 32; idx += 256) {
    const int nn = idx >> 5, o = idx & 31;
    float s = bg[o];
    for (int m = 0; m < 62; ++m) s += sAdj[nn * 62 + m] * sG[m * 32 + o];
    s = s > 0.f ? s : 0.f;
    feats_out[(size_t)b * 5952 + (size_t)(stage * 62 + nn) * 32 + o] = s;
  }
}

// ---------------------------------------------------------------------------
// Final: logits = feats @ w_lin + b_lin; softmax. d_out = [logits | probs]
// ---------------------------------------------------------------------------
__global__ __launch_bounds__(128) void final_kernel(const float* __restrict__ feats,
                                                    const float* __restrict__ wlin,
                                                    const float* __restrict__ blin,
                                                    float* __restrict__ out) {
  const int t = threadIdx.x;  // 128 graphs
  const float* f = feats + (size_t)t * 5952;
  float l0 = blin[0], l1 = blin[1], l2 = blin[2];
  for (int i = 0; i < 5952; ++i) {
    float v = f[i];
    l0 += v * wlin[i * 3 + 0];
    l1 += v * wlin[i * 3 + 1];
    l2 += v * wlin[i * 3 + 2];
  }
  out[t * 3 + 0] = l0; out[t * 3 + 1] = l1; out[t * 3 + 2] = l2;
  float mx = fmaxf(l0, fmaxf(l1, l2));
  float e0 = __expf(l0 - mx), e1 = __expf(l1 - mx), e2 = __expf(l2 - mx);
  float inv = 1.f / (e0 + e1 + e2);
  out[384 + t * 3 + 0] = e0 * inv;
  out[384 + t * 3 + 1] = e1 * inv;
  out[384 + t * 3 + 2] = e2 * inv;
}

// ---------------------------------------------------------------------------
extern "C" void kernel_launch(void* const* d_in, const int* in_sizes, int n_in,
                              void* d_out, int out_size, void* d_ws, size_t ws_size,
                              hipStream_t stream) {
  const float* x    = (const float*)d_in[0];
  const float* wc1  = (const float*)d_in[1];
  const float* bc1  = (const float*)d_in[2];
  const float* wc2  = (const float*)d_in[3];
  const float* bc2  = (const float*)d_in[4];
  const float* wc3  = (const float*)d_in[5];
  const float* bc3  = (const float*)d_in[6];
  const float* wbn1 = (const float*)d_in[7];
  const float* bbn1 = (const float*)d_in[8];
  const float* wg1  = (const float*)d_in[9];
  const float* bg1  = (const float*)d_in[10];
  const float* wbn2 = (const float*)d_in[11];
  const float* bbn2 = (const float*)d_in[12];
  const float* wg2  = (const float*)d_in[13];
  const float* bg2  = (const float*)d_in[14];
  const float* wbn3 = (const float*)d_in[15];
  const float* bbn3 = (const float*)d_in[16];
  const float* wg3  = (const float*)d_in[17];
  const float* bg3  = (const float*)d_in[18];
  const float* wlin = (const float*)d_in[19];
  const float* blin = (const float*)d_in[20];
  float* out = (float*)d_out;

  char* ws = (char*)d_ws;
  size_t off = 0;
  auto carve = [&](size_t bytes) -> void* {
    void* p = ws + off;
    off += (bytes + 255) & ~(size_t)255;
    return p;
  };
  _Float16* A1  = (_Float16*)carve(1024   * 2);
  _Float16* A2  = (_Float16*)carve(10240  * 2);
  _Float16* A3  = (_Float16*)carve(40960  * 2);
  unsigned* BN1 = (unsigned*)carve(133120 * 2);
  unsigned* G1  = (unsigned*)carve(66560  * 2);
  unsigned* BN2 = (unsigned*)carve(61440  * 2);
  unsigned* G2  = (unsigned*)carve(30720  * 2);
  unsigned* BN3 = (unsigned*)carve(16384  * 2);
  unsigned* G3  = (unsigned*)carve(8192   * 2);
  _Float16* h1  = (_Float16*)carve((size_t)7936 * 2080 * 2);
  _Float16* h2  = (_Float16*)carve((size_t)7936 * 960  * 2);
  _Float16* h3  = (_Float16*)carve((size_t)7936 * 256  * 2);
  float* feats  = (float*)carve((size_t)128 * 5952 * 4);

  prep_conv<<<204, 256, 0, stream>>>(wc1, wc2, wc3, A1, A2, A3);
  prep_swz<<<260, 256, 0, stream>>>(wbn1, BN1, 2080, 64);
  prep_swz<<<130, 256, 0, stream>>>(wg1,  G1,  2080, 32);
  prep_swz<<<120, 256, 0, stream>>>(wbn2, BN2, 960, 64);
  prep_swz<<<60,  256, 0, stream>>>(wg2,  G2,  960, 32);
  prep_swz<<<32,  256, 0, stream>>>(wbn3, BN3, 256, 64);
  prep_swz<<<16,  256, 0, stream>>>(wg3,  G3,  256, 32);

  conv1_kernel<<<7936, 256, 0, stream>>>(x, A1, bc1, h1);
  sogc_kernel<2080><<<128, 256, 0, stream>>>(h1, BN1, bbn1, G1, bg1, feats, 0);
  conv2_kernel<<<7936, 256, 0, stream>>>(h1, A2, bc2, h2);
  sogc_kernel<960><<<128, 256, 0, stream>>>(h2, BN2, bbn2, G2, bg2, feats, 1);
  conv3_kernel<<<7936, 256, 0, stream>>>(h2, A3, bc3, h3);
  sogc_kernel<256><<<128, 256, 0, stream>>>(h3, BN3, bbn3, G3, bg3, feats, 2);
  final_kernel<<<1, 128, 0, stream>>>(feats, wlin, blin, out);
}
